// FusedEPMoE_33638183862746
// MI455X (gfx1250) — matrix operand order repaired and checked by reading, not compiled, not run on hardware
//
#include <hip/hip_runtime.h>
#include <hip/hip_bf16.h>
#include <math.h>

// Problem dims (fixed by the reference)
#define T_TOK 1024
#define NEXP  8
#define TOPK  2
#define HID   1024
#define INTER 2048

// 2048 assignments, padded per expert to multiples of 32 -> <= 2048 + 8*31 = 2296
#define MAX_TILES 72                 // 2048/32 + 8
#define MAX_ROWS  (MAX_TILES * 32)   // 2304

typedef __attribute__((ext_vector_type(16))) __bf16 v16bf;
typedef __attribute__((ext_vector_type(8)))  float  v8f;

// ---- workspace layout (bytes) ----
// ctrl ints: [0..7] counts, [8..16] seg_start (9 entries), [17..24] fill, [25] padded_total
#define WS_CTRL  0
#define WS_IDS   128                       // int[T_TOK*TOPK]
#define WS_GATES (WS_IDS + T_TOK*TOPK*4)   // float[T_TOK*TOPK]
#define WS_ATOK  (WS_GATES + T_TOK*TOPK*4) // int[MAX_ROWS]
#define WS_AGATE (WS_ATOK + MAX_ROWS*4)    // float[MAX_ROWS]
#define WS_ABUF  36864                     // __bf16[MAX_ROWS*INTER] (aligned)

// ---------------- kernel 0: zero output + control ----------------
__global__ void k_zero(float* __restrict__ out, int* __restrict__ ctrl, int n) {
  int gid = blockIdx.x * blockDim.x + threadIdx.x;
  for (int i = gid; i < n; i += gridDim.x * blockDim.x) out[i] = 0.0f;
  if (gid < 32) ctrl[gid] = 0;
}

// ---------------- kernel 1: softmax + top-2 router ----------------
__global__ void k_router(const float* __restrict__ logits, int* __restrict__ ctrl,
                         int* __restrict__ ids, float* __restrict__ gates) {
  int t = blockIdx.x * blockDim.x + threadIdx.x;
  if (t >= T_TOK) return;
  float l[NEXP];
  float m = -INFINITY;
#pragma unroll
  for (int e = 0; e < NEXP; ++e) { l[e] = logits[t * NEXP + e]; m = fmaxf(m, l[e]); }
  float p[NEXP]; float s = 0.0f;
#pragma unroll
  for (int e = 0; e < NEXP; ++e) { p[e] = __expf(l[e] - m); s += p[e]; }
  float inv = 1.0f / s;
  int a0 = 0;
#pragma unroll
  for (int e = 1; e < NEXP; ++e) if (l[e] > l[a0]) a0 = e;
  int a1 = (a0 == 0) ? 1 : 0;
#pragma unroll
  for (int e = 0; e < NEXP; ++e) if (e != a0 && l[e] > l[a1]) a1 = e;
  ids[t * 2 + 0] = a0; gates[t * 2 + 0] = p[a0] * inv;
  ids[t * 2 + 1] = a1; gates[t * 2 + 1] = p[a1] * inv;
  atomicAdd(&ctrl[a0], 1);
  atomicAdd(&ctrl[a1], 1);
}

// ---------------- kernel 2: padded exclusive scan + init assignment ----------------
__global__ void k_scan(int* __restrict__ ctrl, int* __restrict__ atok,
                       float* __restrict__ agate) {
  if (threadIdx.x == 0) {
    int run = 0;
    for (int e = 0; e < NEXP; ++e) {
      ctrl[8 + e] = run;
      run += (ctrl[e] + 31) & ~31;   // pad each expert segment to 32-row tiles
      ctrl[17 + e] = 0;              // fill counters
    }
    ctrl[16] = run;                  // end sentinel
    ctrl[25] = run;                  // padded_total
  }
  __syncthreads();
  for (int i = threadIdx.x; i < MAX_ROWS; i += blockDim.x) {
    atok[i] = -1; agate[i] = 0.0f;
  }
}

// ---------------- kernel 3: scatter tokens into expert segments ----------------
__global__ void k_scatter(const int* __restrict__ ids, const float* __restrict__ gates,
                          int* __restrict__ ctrl, int* __restrict__ atok,
                          float* __restrict__ agate) {
  int t = blockIdx.x * blockDim.x + threadIdx.x;
  if (t >= T_TOK) return;
#pragma unroll
  for (int k = 0; k < TOPK; ++k) {
    int e = ids[t * 2 + k];
    int pos = atomicAdd(&ctrl[17 + e], 1);
    int idx = ctrl[8 + e] + pos;
    atok[idx] = t;
    agate[idx] = gates[t * 2 + k];
  }
}

// A fragment (16x32 bf16) from LDS tile rows [r0..r0+15] of a [32][32] tile.
// CDNA5 16-bit A layout: lane<16: M=lane, K halves {0..7,16..23};
//                        lane>=16: M=lane-16, K halves {8..15,24..31}
__device__ __forceinline__ v16bf load_afrag(const __bf16* xt, int r0, int lane) {
  int r = r0 + (lane & 15);
  int kb = (lane < 16) ? 0 : 8;
  v16bf a;
#pragma unroll
  for (int j = 0; j < 8; ++j) {
    a[j]     = xt[r * 32 + kb + j];
    a[8 + j] = xt[r * 32 + kb + 16 + j];
  }
  return a;
}

// B fragment (32x16): lane L holds column N=L%16; K halves {0..7,16..23} (L<16) or {8..15,24..31} (L>=16).
// Source is fp32 row-major [K x ld]; convert to bf16 in-register (no extra HBM traffic).
__device__ __forceinline__ v16bf load_bfrag_f32(const float* __restrict__ w, int ld,
                                                int k0, int n0, int lane) {
  const float* p = w + (size_t)k0 * ld + n0 + (lane & 15);
  int kb = (lane < 16) ? 0 : 8;
  v16bf b;
#pragma unroll
  for (int j = 0; j < 8; ++j) {
    b[j]     = (__bf16)p[(size_t)(kb + j) * ld];
    b[8 + j] = (__bf16)p[(size_t)(kb + 16 + j) * ld];
  }
  return b;
}

// ---------------- kernel 4: grouped GEMM1  a = silu(x@w1) * (x@w3) ----------------
// block: 256 thr (8 waves). Tile: 32 rows (gathered tokens) x 128 cols of I.
// Each wave: one 16-col subtile, two 16-row subtiles sharing the B fragments.
__global__ void __launch_bounds__(256) k_gemm1(
    const float* __restrict__ x, const float* __restrict__ w1,
    const float* __restrict__ w3, const int* __restrict__ ctrl,
    const int* __restrict__ atok, __bf16* __restrict__ abuf) {
  int ptotal = ctrl[25];
  int tile = blockIdx.x;
  if (tile * 32 >= ptotal) return;       // block-uniform exit
  int rowBase = tile * 32;
  int e = 0;
#pragma unroll
  for (int i = 1; i < NEXP; ++i) if (rowBase >= ctrl[8 + i]) e = i;
  const float* w1e = w1 + (size_t)e * HID * INTER;
  const float* w3e = w3 + (size_t)e * HID * INTER;

  __shared__ __bf16 xt[32 * 32];
  __shared__ int stok[32];
  int lane = threadIdx.x & 31;
  int wave = threadIdx.x >> 5;
  int n0 = blockIdx.y * 128 + wave * 16;

  if (threadIdx.x < 32) {
    int tk = atok[rowBase + threadIdx.x];
    stok[threadIdx.x] = (tk < 0) ? 0 : tk;   // padding rows read token 0 (harmless)
  }
  __syncthreads();

  v8f h0 = {}, h1 = {}, u0 = {}, u1 = {};
  for (int k0 = 0; k0 < HID; k0 += 32) {
    // cooperative gather+convert of x tile: 32x32 fp32 -> bf16 LDS, 4 contiguous/thread
    {
      int idx4 = threadIdx.x * 4;           // 0..1020
      int r = idx4 >> 5, c = idx4 & 31;
      const float* xp = x + (size_t)stok[r] * HID + k0 + c;
#pragma unroll
      for (int j = 0; j < 4; ++j) xt[idx4 + j] = (__bf16)xp[j];
    }
    __syncthreads();
    v16bf a0 = load_afrag(xt, 0, lane);
    v16bf a1 = load_afrag(xt, 16, lane);
    v16bf b1 = load_bfrag_f32(w1e, INTER, k0, n0, lane);
    v16bf b3 = load_bfrag_f32(w3e, INTER, k0, n0, lane);
    h0 = __builtin_amdgcn_wmma_f32_16x16x32_bf16(false, a0, false, b1, (short)0, h0, false, false);
    h1 = __builtin_amdgcn_wmma_f32_16x16x32_bf16(false, a1, false, b1, (short)0, h1, false, false);
    u0 = __builtin_amdgcn_wmma_f32_16x16x32_bf16(false, a0, false, b3, (short)0, u0, false, false);
    u1 = __builtin_amdgcn_wmma_f32_16x16x32_bf16(false, a1, false, b3, (short)0, u1, false, false);
    __syncthreads();
  }

  // epilogue: silu(h)*u via v_rcp_f32, store bf16 row-major (lanes 0..15 contiguous in N)
  int n = lane & 15;
  int moff = (lane < 16) ? 0 : 8;
#pragma unroll
  for (int v = 0; v < 8; ++v) {
    float ha = h0[v], ua = u0[v];
    float hb = h1[v], ub = u1[v];
    float sa = __builtin_amdgcn_rcpf(1.0f + __expf(-ha));
    float sb = __builtin_amdgcn_rcpf(1.0f + __expf(-hb));
    abuf[(size_t)(rowBase + moff + v) * INTER + n0 + n]      = (__bf16)(ha * sa * ua);
    abuf[(size_t)(rowBase + 16 + moff + v) * INTER + n0 + n] = (__bf16)(hb * sb * ub);
  }
}

// ---------------- kernel 5: grouped GEMM2  out[t] += gate * (a @ w2) ----------------
// block: 256 thr (8 waves). Tile: 32 rows x 128 cols of H; B fragment shared by 2 A frags.
__global__ void __launch_bounds__(256) k_gemm2(
    const float* __restrict__ w2, const int* __restrict__ ctrl,
    const int* __restrict__ atok, const float* __restrict__ agate,
    const __bf16* __restrict__ abuf, float* __restrict__ out) {
  int ptotal = ctrl[25];
  int tile = blockIdx.x;
  if (tile * 32 >= ptotal) return;
  int rowBase = tile * 32;
  int e = 0;
#pragma unroll
  for (int i = 1; i < NEXP; ++i) if (rowBase >= ctrl[8 + i]) e = i;
  const float* w2e = w2 + (size_t)e * INTER * HID;

  __shared__ __bf16 at[32 * 32];
  __shared__ int stok[32];
  __shared__ float sg[32];
  int lane = threadIdx.x & 31;
  int wave = threadIdx.x >> 5;
  int n0 = blockIdx.y * 128 + wave * 16;

  if (threadIdx.x < 32) {
    int tk = atok[rowBase + threadIdx.x];
    stok[threadIdx.x] = (tk < 0) ? 0 : tk;
    sg[threadIdx.x] = (tk < 0) ? 0.0f : agate[rowBase + threadIdx.x];
  }
  __syncthreads();

  v8f acc0 = {}, acc1 = {};
  for (int k0 = 0; k0 < INTER; k0 += 32) {
    {
      int idx4 = threadIdx.x * 4;
      int r = idx4 >> 5, c = idx4 & 31;
      const __bf16* ap = abuf + (size_t)(rowBase + r) * INTER + k0 + c;
#pragma unroll
      for (int j = 0; j < 4; ++j) at[idx4 + j] = ap[j];
    }
    __syncthreads();
    v16bf a0 = load_afrag(at, 0, lane);
    v16bf a1 = load_afrag(at, 16, lane);
    v16bf bf = load_bfrag_f32(w2e, HID, k0, n0, lane);
    acc0 = __builtin_amdgcn_wmma_f32_16x16x32_bf16(false, a0, false, bf, (short)0, acc0, false, false);
    acc1 = __builtin_amdgcn_wmma_f32_16x16x32_bf16(false, a1, false, bf, (short)0, acc1, false, false);
    __syncthreads();
  }

  int n = lane & 15;
  int moff = (lane < 16) ? 0 : 8;
#pragma unroll
  for (int v = 0; v < 8; ++v) {
    int m0 = moff + v;
    int m1 = 16 + moff + v;
    atomicAdd(&out[(size_t)stok[m0] * HID + n0 + n], sg[m0] * acc0[v]);
    atomicAdd(&out[(size_t)stok[m1] * HID + n0 + n], sg[m1] * acc1[v]);
  }
}

extern "C" void kernel_launch(void* const* d_in, const int* in_sizes, int n_in,
                              void* d_out, int out_size, void* d_ws, size_t ws_size,
                              hipStream_t stream) {
  const float* x      = (const float*)d_in[0];
  const float* logits = (const float*)d_in[1];
  const float* w1     = (const float*)d_in[2];
  const float* w3     = (const float*)d_in[3];
  const float* w2     = (const float*)d_in[4];
  float* out = (float*)d_out;

  char* ws = (char*)d_ws;
  int*    ctrl  = (int*)(ws + WS_CTRL);
  int*    ids   = (int*)(ws + WS_IDS);
  float*  gates = (float*)(ws + WS_GATES);
  int*    atok  = (int*)(ws + WS_ATOK);
  float*  agate = (float*)(ws + WS_AGATE);
  __bf16* abuf  = (__bf16*)(ws + WS_ABUF);

  k_zero<<<256, 256, 0, stream>>>(out, ctrl, T_TOK * HID);
  k_router<<<(T_TOK + 255) / 256, 256, 0, stream>>>(logits, ctrl, ids, gates);
  k_scan<<<1, 256, 0, stream>>>(ctrl, atok, agate);
  k_scatter<<<(T_TOK + 255) / 256, 256, 0, stream>>>(ids, gates, ctrl, atok, agate);

  dim3 g1(MAX_TILES, INTER / 128);
  k_gemm1<<<g1, 256, 0, stream>>>(x, w1, w3, ctrl, atok, abuf);
  dim3 g2(MAX_TILES, HID / 128);
  k_gemm2<<<g2, 256, 0, stream>>>(w2, ctrl, atok, agate, abuf, out);
}
